// ChunkedSelfAttention_1503238554306
// MI455X (gfx1250) — matrix operand order repaired and checked by reading, not compiled, not run on hardware
//
#include <hip/hip_runtime.h>
#include <hip/hip_bf16.h>
#include <math.h>

typedef __attribute__((ext_vector_type(16))) _Float16 v16h;
typedef __attribute__((ext_vector_type(8)))  float    v8f;
typedef __attribute__((ext_vector_type(4)))  unsigned uv4;

union frag16 {
    v16h v;
    unsigned u[8];
    uv4 q[2];
};

#define BB 2
#define TT 4096
#define HH 16
#define DH 128
#define DV 128
#define CH 2048
#define NC (TT / CH)          // 2 chunks
#define HALF_D (DH / 2)       // 64 (RoPE half)

#define KPAD 136              // K tile row pitch in halves (272B = 68 dwords)
#define VPAD 72               // V tile row pitch in halves (144B = 36 dwords)
#define PPAD 72               // P scratch row pitch in halves

// ---------------------------------------------------------------------------
// Kernel 1: RoPE on q,k (f32 math, matches reference), convert to f16,
// relayout [B,T,H,D] -> [B,H,T,D] so attention rows are contiguous.
// ---------------------------------------------------------------------------
__global__ __launch_bounds__(256) void rope_qk_kernel(
    const float* __restrict__ q, const float* __restrict__ k,
    const int* __restrict__ start_index,
    _Float16* __restrict__ qh, _Float16* __restrict__ kh)
{
    unsigned tid = blockIdx.x * 256u + threadIdx.x;   // 2^23 threads total
    int j = tid & 63;
    int h = (tid >> 6) & 15;
    int t = (tid >> 10) & 4095;
    int b = tid >> 22;

    float inv = __expf(-(logf(10000.0f) / (float)HALF_D) * (float)j);
    float pos = (float)t + (float)(*start_index);
    float s, c;
    __sincosf(pos * inv, &s, &c);

    size_t in_i  = ((size_t)(b * TT + t) * HH + h) * DH + j;
    size_t out_i = ((size_t)(b * HH + h) * TT + t) * DH + j;

    float q1 = q[in_i], q2 = q[in_i + HALF_D];
    qh[out_i]          = (_Float16)(q1 * c - q2 * s);
    qh[out_i + HALF_D] = (_Float16)(q2 * c + q1 * s);

    float k1 = k[in_i], k2 = k[in_i + HALF_D];
    kh[out_i]          = (_Float16)(k1 * c - k2 * s);
    kh[out_i + HALF_D] = (_Float16)(k2 * c + k1 * s);
}

// ---------------------------------------------------------------------------
// Kernel 2: convert V to f16 and transpose to [B,H,DV,T] via LDS 32x32 tile
// so PV B-fragments are contiguous along the K(kv) dimension.
// ---------------------------------------------------------------------------
__global__ __launch_bounds__(256) void v_transpose_kernel(
    const float* __restrict__ v, _Float16* __restrict__ vt)
{
    __shared__ float tile[32][33];
    unsigned bid = blockIdx.x;                 // 2^14 blocks
    int tq = bid & 127;
    int dq = (bid >> 7) & 3;
    int h  = (bid >> 9) & 15;
    int b  = bid >> 13;
    int tx = threadIdx.x & 31;
    int ty = threadIdx.x >> 5;
    int t0 = tq * 32, d0 = dq * 32;

    #pragma unroll
    for (int i = 0; i < 4; ++i) {
        int t = t0 + ty + 8 * i;
        tile[ty + 8 * i][tx] = v[((size_t)(b * TT + t) * HH + h) * DV + d0 + tx];
    }
    __syncthreads();
    #pragma unroll
    for (int i = 0; i < 4; ++i) {
        int d = d0 + ty + 8 * i;
        vt[((size_t)(b * HH + h) * DV + d) * TT + t0 + tx] = (_Float16)tile[tx][ty + 8 * i];
    }
}

// ---------------------------------------------------------------------------
// Kernel 3: flash attention. Block = 8 waves sharing one (b,h,ch) slice and
// 128 q rows; K/V kv-tiles (64 wide) staged in LDS and shared by all 8 waves
// (128 FLOP per L2 byte). Register double-buffering overlaps the global loads
// of tile i+1 with the WMMA compute of tile i.
// ---------------------------------------------------------------------------
__global__ __launch_bounds__(256) void attn_kernel(
    const _Float16* __restrict__ qh, const _Float16* __restrict__ kh,
    const _Float16* __restrict__ vt, float* __restrict__ out)
{
    __shared__ _Float16 k_sm[64][KPAD];        // 17408 B
    __shared__ _Float16 v_sm[128][VPAD];       // 18432 B
    __shared__ _Float16 p_sm[8][16][PPAD];     // 18432 B   (total 53 KB)

    const int lane = threadIdx.x & 31;
    const int w    = threadIdx.x >> 5;
    const int l15  = lane & 15;
    const int hi   = lane >> 4;

    // block decode: 1024 blocks = B * H * NC * (CH/128)
    const int blk = blockIdx.x;
    const int bq  = blk & 15;                  // 128-row block within chunk
    const int ch  = (blk >> 4) & (NC - 1);
    const int h   = (blk >> 5) & 15;
    const int b   = blk >> 9;

    const int r0 = bq * 128 + w * 16;          // this wave's first q row
    const int tq = ch * CH + r0;

    const _Float16* qbase = qh + (size_t)(b * HH + h) * TT * DH;
    const _Float16* kbase = kh + (size_t)(b * HH + h) * TT * DH;
    const _Float16* vbase = vt + (size_t)(b * HH + h) * DV * TT;

    // ---- load Q A-fragments (16x32 each, Dh=128 -> 4 fragments) ----
    frag16 aq[4];
    {
        const unsigned* qrow = (const unsigned*)(qbase + (size_t)(tq + l15) * DH);
        #pragma unroll
        for (int df = 0; df < 4; ++df)
            #pragma unroll
            for (int p = 0; p < 8; ++p) {
                int koff = 2 * p + hi * 16;
                aq[df].u[p] = qrow[(32 * df + koff) >> 1];
            }
    }

    float m_r[8], l_r[8];
    v8f acc[8] = {};
    #pragma unroll
    for (int vv = 0; vv < 8; ++vv) { m_r[vv] = -3.0e38f; l_r[vv] = 0.0f; }

    const int ntiles = 2 * bq + 2;             // block-uniform kv tile count

    // ---- staging: register double buffer for K/V tile ----
    uv4 kreg[2], vreg[2];
    auto load_tile = [&](int kv0) {
        #pragma unroll
        for (int i = 0; i < 2; ++i) {
            int item = (int)threadIdx.x + i * 256;    // 0..511
            int kr = item >> 3, kc = item & 7;        // K: 64 rows x 8 uv4
            kreg[i] = *(const uv4*)(kbase + (size_t)(ch * CH + kv0 + kr) * DH + kc * 8);
            int vr = item >> 2, vc = item & 3;        // V: 128 rows x 4 uv4
            vreg[i] = *(const uv4*)(vbase + (size_t)vr * TT + (ch * CH + kv0 + vc * 8));
        }
    };
    auto store_tile = [&]() {
        #pragma unroll
        for (int i = 0; i < 2; ++i) {
            int item = (int)threadIdx.x + i * 256;
            int kr = item >> 3, kc = item & 7;
            *(uv4*)&k_sm[kr][kc * 8] = kreg[i];
            int vr = item >> 2, vc = item & 3;
            *(uv4*)&v_sm[vr][vc * 8] = vreg[i];
        }
    };

    load_tile(0);

    for (int it = 0; it < ntiles; ++it) {
        const int kv0 = it * 64;

        __syncthreads();                       // previous tile consumed
        store_tile();
        if (it + 1 < ntiles) load_tile(kv0 + 64);   // overlaps compute below
        __syncthreads();                       // tile ready

        // tiles entirely above this wave's rows are an exact no-op
        if (kv0 <= r0 + 15) {
            // ---- S = Q * K^T : 16 WMMAs, double-buffered LDS B frags ----
            v8f s[4] = {};
            frag16 bk[2];
            auto loadK = [&](frag16& f, int idx) {
                int nf = idx >> 2, df = idx & 3;
                const uv4* p = (const uv4*)&k_sm[16 * nf + l15][32 * df + 16 * hi];
                f.q[0] = p[0];
                f.q[1] = p[1];
            };
            loadK(bk[0], 0);
            #pragma unroll
            for (int idx = 0; idx < 16; ++idx) {
                if (idx < 15) loadK(bk[(idx + 1) & 1], idx + 1);
                int nf = idx >> 2;
                s[nf] = __builtin_amdgcn_wmma_f32_16x16x32_f16(
                    false, aq[idx & 3].v, false, bk[idx & 1].v, (short)0, s[nf],
                    false, false);
            }

            // ---- causal mask ----
            if (kv0 + 63 > r0) {
                #pragma unroll
                for (int nf = 0; nf < 4; ++nf)
                    #pragma unroll
                    for (int vv = 0; vv < 8; ++vv) {
                        int col = kv0 + 16 * nf + l15;
                        int row = r0 + vv + 8 * hi;
                        s[nf][vv] = (col > row) ? -1.0e30f : s[nf][vv];
                    }
            }

            // ---- online softmax ----
            float mt[8];
            #pragma unroll
            for (int vv = 0; vv < 8; ++vv) {
                float t = fmaxf(fmaxf(s[0][vv], s[1][vv]), fmaxf(s[2][vv], s[3][vv]));
                t = fmaxf(t, __shfl_xor(t, 1, 32));
                t = fmaxf(t, __shfl_xor(t, 2, 32));
                t = fmaxf(t, __shfl_xor(t, 4, 32));
                t = fmaxf(t, __shfl_xor(t, 8, 32));
                mt[vv] = t;
            }
            float alpha[8], rs[8];
            #pragma unroll
            for (int vv = 0; vv < 8; ++vv) {
                float mn = fmaxf(m_r[vv], mt[vv]);
                alpha[vv] = __expf(m_r[vv] - mn);
                m_r[vv] = mn;
                rs[vv] = 0.0f;
            }

            // ---- P = exp(S-m) -> per-wave LDS (C-layout -> row major) ----
            #pragma unroll
            for (int nf = 0; nf < 4; ++nf)
                #pragma unroll
                for (int vv = 0; vv < 8; ++vv) {
                    float p = __expf(s[nf][vv] - m_r[vv]);
                    rs[vv] += p;
                    p_sm[w][vv + 8 * hi][16 * nf + l15] = (_Float16)p;
                }
            #pragma unroll
            for (int vv = 0; vv < 8; ++vv) {
                float t = rs[vv];
                t += __shfl_xor(t, 1, 32);
                t += __shfl_xor(t, 2, 32);
                t += __shfl_xor(t, 4, 32);
                t += __shfl_xor(t, 8, 32);
                l_r[vv] = l_r[vv] * alpha[vv] + t;
            }
            #pragma unroll
            for (int g = 0; g < 8; ++g)
                #pragma unroll
                for (int vv = 0; vv < 8; ++vv)
                    acc[g][vv] *= alpha[vv];

            // ---- reload P as A-fragments (same-wave LDS is in-order) ----
            frag16 ap[2];
            {
                const unsigned* prow = (const unsigned*)&p_sm[w][l15][0];
                #pragma unroll
                for (int kf = 0; kf < 2; ++kf)
                    #pragma unroll
                    for (int p = 0; p < 8; ++p) {
                        int koff = 2 * p + hi * 16;
                        ap[kf].u[p] = prow[(32 * kf + koff) >> 1];
                    }
            }

            // ---- acc += P * V : 16 WMMAs, double-buffered LDS B frags ----
            frag16 bv[2];
            auto loadV = [&](frag16& f, int idx) {
                int g = idx >> 1, kf = idx & 1;
                const uv4* p = (const uv4*)&v_sm[16 * g + l15][32 * kf + 16 * hi];
                f.q[0] = p[0];
                f.q[1] = p[1];
            };
            loadV(bv[0], 0);
            #pragma unroll
            for (int idx = 0; idx < 16; ++idx) {
                if (idx < 15) loadV(bv[(idx + 1) & 1], idx + 1);
                int g = idx >> 1;
                acc[g] = __builtin_amdgcn_wmma_f32_16x16x32_f16(
                    false, ap[idx & 1].v, false, bv[idx & 1].v, (short)0, acc[g],
                    false, false);
            }
        }
    }

    // ---- normalize and store: out[b, t, h, dv] f32 ----
    #pragma unroll
    for (int g = 0; g < 8; ++g)
        #pragma unroll
        for (int vv = 0; vv < 8; ++vv) {
            int row = r0 + vv + 8 * hi;
            int t = ch * CH + row;
            out[((size_t)(b * TT + t) * HH + h) * DV + 16 * g + l15] =
                acc[g][vv] / l_r[vv];
        }
}

// ---------------------------------------------------------------------------
extern "C" void kernel_launch(void* const* d_in, const int* in_sizes, int n_in,
                              void* d_out, int out_size, void* d_ws, size_t ws_size,
                              hipStream_t stream) {
    const float* q = (const float*)d_in[0];
    const float* k = (const float*)d_in[1];
    const float* v = (const float*)d_in[2];
    const int* start_index = (const int*)d_in[3];
    float* out = (float*)d_out;

    const size_t E = (size_t)BB * HH * TT * DH;    // 16,777,216 elems
    _Float16* qh = (_Float16*)d_ws;
    _Float16* kh = qh + E;
    _Float16* vt = kh + E;

    rope_qk_kernel<<<(BB * TT * HH * HALF_D) / 256, 256, 0, stream>>>(
        q, k, start_index, qh, kh);

    v_transpose_kernel<<<BB * HH * (TT / 32) * (DV / 32), 256, 0, stream>>>(v, vt);

    attn_kernel<<<BB * HH * NC * (CH / 128), 256, 0, stream>>>(qh, kh, vt, out);
}
